// VectorQuantizerWithKld_11166914969843
// MI455X (gfx1250) — compile-verified, matrix operand order checked by reading, not run on hardware
//
#include <hip/hip_runtime.h>
#include <hip/hip_bf16.h>

// ---------------------------------------------------------------------------
// VQ quantizer for MI455X (gfx1250), wave32, WMMA bf16 path.
//
// Shapes: z [16,64,64,64] fp32 (B,C,H,W), codebook [2048,64] fp32.
// N = 65536 pixels, K = 64 channels, n_e = 2048 codes.
// Outputs (flat in d_out): z_q [4194304] f32, loss [1], perplexity [1],
// indices [65536] int32 (bit-stored in the float buffer tail).
//
// Fused logits-GEMM + argmax never materializes the 512 MB logits matrix.
// Codes sit on the WMMA M axis, pixels on N, so each lane's 8 accumulator
// floats are 8 code-logits of ONE pixel -> in-lane running argmax + one
// SWAPX16 ds_swizzle combine at the end.
// ---------------------------------------------------------------------------

typedef __attribute__((ext_vector_type(16))) __bf16 v16bf;
typedef __attribute__((ext_vector_type(8)))  float  v8f;
typedef int vsi4 __attribute__((__vector_size__(16)));   // builtin's int4 type

#define NPIX   65536      // B*H*W
#define NCODE  2048
#define KDIM   64
#define CHUNK  512        // codes staged per LDS round
#define LDS_ROW_BYTES 144 // 128B row + 16B pad -> conflict-free 16-lane reads
#define LDS_BYTES (CHUNK * LDS_ROW_BYTES)   // 73728

// ---- ws layout (bytes) ----
#define WS_ZT     0                    // bf16 z transposed [N][64]   : 8 MB
#define WS_CB16   8388608              // bf16 codebook [2048][64]    : 256 KB
#define WS_HIST   8650752              // uint32[2048]
#define WS_LOSS   8658944              // float[1]

#define ASG __attribute__((address_space(1)))
#define ASL __attribute__((address_space(3)))

union BF16Frag { v16bf v; uint4 q[2]; };

__device__ __forceinline__ void wait_asynccnt0() {
#if __has_builtin(__builtin_amdgcn_s_wait_asynccnt)
    __builtin_amdgcn_s_wait_asynccnt(0);
#else
    asm volatile("s_wait_asynccnt 0x0" ::: "memory");
#endif
}

// ---------------------------------------------------------------------------
__global__ void vq_zero_kernel(unsigned* hist, float* lossAcc) {
    int t = blockIdx.x * blockDim.x + threadIdx.x;
    if (t < NCODE) hist[t] = 0u;
    if (t == 0)    *lossAcc = 0.0f;
}

// codebook fp32 -> bf16 row-major [2048][64]
__global__ void vq_cvt_cb_kernel(const float* __restrict__ cb,
                                 __bf16* __restrict__ cb16) {
    int t = blockIdx.x * blockDim.x + threadIdx.x;   // 131072 elems
    cb16[t] = (__bf16)cb[t];
}

// z [B,C,HW] fp32 -> zt [N=B*HW][64] bf16 (pixel-major rows for fragments)
__global__ void vq_cvt_z_kernel(const float* __restrict__ z,
                                __bf16* __restrict__ zt) {
    int t = blockIdx.x * blockDim.x + threadIdx.x;   // 4194304, coalesced read
    int b  = t >> 18;          // / (64*4096)
    int c  = (t >> 12) & 63;
    int hw = t & 4095;
    int n  = (b << 12) | hw;
    zt[(size_t)n * KDIM + c] = (__bf16)z[t];
}

// ---------------------------------------------------------------------------
// Fused logits GEMM + argmax. Per wave: 32 pixels (2 B-tiles), all 2048
// codes in 16-code tiles; codebook chunk staged in LDS by the whole block
// via async global->LDS copies (ASYNCcnt-tracked, no VGPR bounce).
// ---------------------------------------------------------------------------
__global__ void __launch_bounds__(256)
vq_argmax_kernel(const __bf16* __restrict__ zt,
                 const __bf16* __restrict__ cb16,
                 int* __restrict__ out_idx) {
    extern __shared__ char smem[];

    const int lane  = threadIdx.x & 31;
    const int wv    = threadIdx.x >> 5;          // 8 waves / block
    const int l15   = lane & 15;
    const int half  = lane >> 4;                 // 0: codes r, 1: codes 8+r
    const int k0    = half << 3;                 // K-phase of this half
    const int pbase = blockIdx.x * 256 + wv * 32;
    const int pix0  = pbase + l15;
    const int pix1  = pbase + 16 + l15;

    // B fragments (pixels): resident in registers for the whole kernel.
    const __bf16* zr0 = zt + (size_t)pix0 * KDIM;
    const __bf16* zr1 = zt + (size_t)pix1 * KDIM;
    BF16Frag b0lo, b0hi, b1lo, b1hi;
    b0lo.q[0] = *(const uint4*)(zr0 + k0);        // K k0..k0+7
    b0lo.q[1] = *(const uint4*)(zr0 + k0 + 16);   // K k0+16..k0+23
    b0hi.q[0] = *(const uint4*)(zr0 + k0 + 32);
    b0hi.q[1] = *(const uint4*)(zr0 + k0 + 48);
    b1lo.q[0] = *(const uint4*)(zr1 + k0);
    b1lo.q[1] = *(const uint4*)(zr1 + k0 + 16);
    b1hi.q[0] = *(const uint4*)(zr1 + k0 + 32);
    b1hi.q[1] = *(const uint4*)(zr1 + k0 + 48);

    float bv0 = -3.0e38f, bv1 = -3.0e38f;
    int   bi0 = 0,        bi1 = 0;

    for (int chunk = 0; chunk < NCODE / CHUNK; ++chunk) {
        const __bf16* src = cb16 + (size_t)chunk * CHUNK * KDIM;

        // cooperative stage: 512 rows x 64B (first 32 k of each row), padded to 144B rows in LDS
#if __has_builtin(__builtin_amdgcn_global_load_async_to_lds_b128)
        for (int u = threadIdx.x; u < CHUNK * 4; u += 256) {
            int r = u >> 2, o = u & 3;
            __builtin_amdgcn_global_load_async_to_lds_b128(
                (ASG vsi4*)(src + r * KDIM + o * 8),
                (ASL vsi4*)(smem + r * LDS_ROW_BYTES + o * 16),
                /*offset=*/0, /*cpol=*/0);
        }
        wait_asynccnt0();
#else
        for (int u = threadIdx.x; u < CHUNK * 4; u += 256) {
            int r = u >> 2, o = u & 3;
            *(uint4*)(smem + r * LDS_ROW_BYTES + o * 16) =
                *(const uint4*)(src + r * KDIM + o * 8);
        }
#endif
        // hint next chunk toward L2 while we crunch this one
        if (chunk + 1 < NCODE / CHUNK)
            __builtin_prefetch(src + CHUNK * KDIM + (threadIdx.x << 6), 0, 1);
        __syncthreads();

        for (int t = 0; t < CHUNK / 16; ++t) {
            const char* arow = smem + (size_t)(t * 16 + l15) * LDS_ROW_BYTES;
            BF16Frag alo, ahi;
            alo.q[0] = *(const uint4*)(arow + k0 * 2);        // K k0..k0+7
            alo.q[1] = *(const uint4*)(arow + k0 * 2 + 32);   // K k0+16..k0+23
            ahi.q[0] = *(const uint4*)(arow + 64 + k0 * 2);   // +32 K
            ahi.q[1] = *(const uint4*)(arow + 64 + k0 * 2 + 32);

            // one A tile feeds 4 WMMAs (2 pixel tiles x K-split)
            v8f c0 = {}, c1 = {};
            c0 = __builtin_amdgcn_wmma_f32_16x16x32_bf16(
                     false, alo.v, false, b0lo.v, (short)0, c0, false, false);
            c0 = __builtin_amdgcn_wmma_f32_16x16x32_bf16(
                     false, ahi.v, false, b0hi.v, (short)0, c0, false, false);
            c1 = __builtin_amdgcn_wmma_f32_16x16x32_bf16(
                     false, alo.v, false, b1lo.v, (short)0, c1, false, false);
            c1 = __builtin_amdgcn_wmma_f32_16x16x32_bf16(
                     false, ahi.v, false, b1hi.v, (short)0, c1, false, false);

            const int codebase = chunk * CHUNK + t * 16 + (half << 3);
#pragma unroll
            for (int r = 0; r < 8; ++r) {
                float v0 = c0[r];
                if (v0 > bv0) { bv0 = v0; bi0 = codebase + r; }
                float v1 = c1[r];
                if (v1 > bv1) { bv1 = v1; bi1 = codebase + r; }
            }
        }
        __syncthreads();
    }

    // combine lane L (codes m%16<8) with lane L^16 (codes m%16>=8): SWAPX16
    float ov0 = __int_as_float(
        __builtin_amdgcn_ds_swizzle(__float_as_int(bv0), 0x401f));
    int   oi0 = __builtin_amdgcn_ds_swizzle(bi0, 0x401f);
    if (ov0 > bv0 || (ov0 == bv0 && oi0 < bi0)) { bv0 = ov0; bi0 = oi0; }
    float ov1 = __int_as_float(
        __builtin_amdgcn_ds_swizzle(__float_as_int(bv1), 0x401f));
    int   oi1 = __builtin_amdgcn_ds_swizzle(bi1, 0x401f);
    if (ov1 > bv1 || (ov1 == bv1 && oi1 < bi1)) { bv1 = ov1; bi1 = oi1; }

    if (lane < 16) {
        out_idx[pix0] = bi0;
        out_idx[pix1] = bi1;
    }
}

// ---------------------------------------------------------------------------
// z_q gather (fp32, NCHW layout), loss partial sum, histogram.
// One thread per pixel; per-channel stores are coalesced across threads.
// ---------------------------------------------------------------------------
__global__ void __launch_bounds__(256)
vq_gather_kernel(const float* __restrict__ z,
                 const float* __restrict__ cb,
                 const int* __restrict__ idx,
                 float* __restrict__ zq,
                 float* __restrict__ lossAcc,
                 unsigned* __restrict__ hist) {
    int n  = blockIdx.x * blockDim.x + threadIdx.x;  // 0..65535
    int b  = n >> 12;
    int hw = n & 4095;
    int i  = idx[n];
    atomicAdd(&hist[i], 1u);

    const float* crow = cb + (size_t)i * KDIM;
    float s = 0.0f;
#pragma unroll 4
    for (int c = 0; c < KDIM; ++c) {
        size_t off = ((size_t)(b * 64 + c) << 12) + hw;
        float q  = crow[c];
        float zv = z[off];
        zq[off]  = q;
        float d  = q - zv;
        s += d * d;
    }

    __shared__ float red[256];
    red[threadIdx.x] = s;
    __syncthreads();
    for (int st = 128; st > 0; st >>= 1) {
        if (threadIdx.x < st) red[threadIdx.x] += red[threadIdx.x + st];
        __syncthreads();
    }
    if (threadIdx.x == 0) atomicAdd(lossAcc, red[0]);
}

// ---------------------------------------------------------------------------
__global__ void vq_final_kernel(const unsigned* __restrict__ hist,
                                const float* __restrict__ lossAcc,
                                float* __restrict__ out_loss,
                                float* __restrict__ out_perp) {
    __shared__ float red[256];
    float s = 0.0f;
    for (int i = threadIdx.x; i < NCODE; i += 256) {
        float em = (float)hist[i] * (1.0f / (float)NPIX);
        s += em * __logf(em + 1e-10f);
    }
    red[threadIdx.x] = s;
    __syncthreads();
    for (int st = 128; st > 0; st >>= 1) {
        if (threadIdx.x < st) red[threadIdx.x] += red[threadIdx.x + st];
        __syncthreads();
    }
    if (threadIdx.x == 0) {
        *out_perp = __expf(-red[0]);
        // forward value of beta*mse(sg(zq),z) + mse(zq,sg(z)) = 1.25*mse
        *out_loss = 1.25f * (*lossAcc) * (1.0f / 4194304.0f);
    }
}

// ---------------------------------------------------------------------------
extern "C" void kernel_launch(void* const* d_in, const int* in_sizes, int n_in,
                              void* d_out, int out_size, void* d_ws, size_t ws_size,
                              hipStream_t stream) {
    const float* z  = (const float*)d_in[0];   // [16,64,64,64]
    const float* cb = (const float*)d_in[1];   // [2048,64]

    float* out      = (float*)d_out;
    float* zq       = out;                     // 4194304
    float* out_loss = out + 4194304;
    float* out_perp = out + 4194305;
    int*   out_idx  = (int*)(out + 4194306);   // 65536 int32

    char*     ws      = (char*)d_ws;
    __bf16*   zt      = (__bf16*)(ws + WS_ZT);
    __bf16*   cb16    = (__bf16*)(ws + WS_CB16);
    unsigned* hist    = (unsigned*)(ws + WS_HIST);
    float*    lossAcc = (float*)(ws + WS_LOSS);

    vq_zero_kernel<<<8, 256, 0, stream>>>(hist, lossAcc);
    vq_cvt_cb_kernel<<<(NCODE * KDIM) / 256, 256, 0, stream>>>(cb, cb16);
    vq_cvt_z_kernel<<<(NPIX * KDIM) / 256, 256, 0, stream>>>(z, zt);
    vq_argmax_kernel<<<NPIX / 256, 256, LDS_BYTES, stream>>>(zt, cb16, out_idx);
    vq_gather_kernel<<<NPIX / 256, 256, 0, stream>>>(z, cb, out_idx, zq, lossAcc, hist);
    vq_final_kernel<<<1, 256, 0, stream>>>(hist, lossAcc, out_loss, out_perp);
}
